// SwinLikeBlock_36558761624098
// MI455X (gfx1250) — compile-verified, matrix operand order checked by reading, not run on hardware
//
#include <hip/hip_runtime.h>
#include <hip/hip_bf16.h>

// ---------------------------------------------------------------------------
// SwinLikeBlock on gfx1250 (MI455X): bf16 WMMA (16x16x32, fp32 accum) for all
// GEMMs + transposed flash attention (softmax stats live on the lane axis, P
// fragment packed directly in registers). GEMM W-tiles staged to LDS with
// global_load_async_to_lds_b128. wave32 everywhere.
// ---------------------------------------------------------------------------

typedef __attribute__((ext_vector_type(16))) __bf16 v16bf;
typedef __attribute__((ext_vector_type(8)))  float  v8f;

#define DIM        384
#define HEADS      8
#define HEAD_DIM   48
#define HPAD       64
#define HIDDEN     1536
#define NTOK       1024        // 32*32 per image
#define BATCH      8
#define MTOT       (BATCH * NTOK)   // 8192 rows
#define SCALE_F    0.14433756729740643f   // 48^-0.5
#define LN_EPS     1e-5f

__device__ __forceinline__ v8f v8f_zero() {
  v8f z = {0.f, 0.f, 0.f, 0.f, 0.f, 0.f, 0.f, 0.f};
  return z;
}

__device__ __forceinline__ v8f wmma_bf16(v16bf a, v16bf b, v8f c) {
  return __builtin_amdgcn_wmma_f32_16x16x32_bf16(
      /*neg_a=*/false, a, /*neg_b=*/false, b,
      /*c_mod=*/(short)0, c, /*reuse_a=*/false, /*reuse_b=*/false);
}

// Load a 16x32 bf16 WMMA A/B fragment from a row-major matrix (ld in elems).
// Per ISA layout: lane l holds row/col (idx0 + l%16); K chunks are
// {base..base+7} and {base+16..base+23} with base = (l/16)*8.
__device__ __forceinline__ v16bf load_frag(const __bf16* __restrict__ base,
                                           int ld, int idx0, int k0) {
  const int lane = threadIdx.x & 31;
  const __bf16* p = base + (size_t)(idx0 + (lane & 15)) * (size_t)ld
                         + (size_t)(k0 + ((lane >> 4) << 3));
  v16bf f;
  uint4* fu = reinterpret_cast<uint4*>(&f);
  fu[0] = *reinterpret_cast<const uint4*>(p);
  fu[1] = *reinterpret_cast<const uint4*>(p + 16);
  return f;
}

// ---------------------------------------------------------------------------
// Elementwise helpers
// ---------------------------------------------------------------------------
__global__ void __launch_bounds__(256) cvt_bf16_k(const float* __restrict__ in,
                                                  __bf16* __restrict__ out, int n) {
  int i = (int)blockIdx.x * 256 + (int)threadIdx.x;
  if (i < n) out[i] = (__bf16)in[i];
}

__global__ void __launch_bounds__(256) zero16_k(uint4* __restrict__ p) {
  size_t i = (size_t)blockIdx.x * 256 + threadIdx.x;
  p[i] = make_uint4(0u, 0u, 0u, 0u);
}

// ---------------------------------------------------------------------------
// LN1 (+ NCHW -> [token, C] transpose): one wave per token.
// ---------------------------------------------------------------------------
__global__ void __launch_bounds__(256) prep_ln1_k(const float* __restrict__ x,
                                                  const float* __restrict__ g,
                                                  const float* __restrict__ be,
                                                  float* __restrict__ xf,
                                                  __bf16* __restrict__ xn) {
  const int tok  = (int)blockIdx.x * 8 + ((int)threadIdx.x >> 5);
  const int lane = threadIdx.x & 31;
  const int b = tok >> 10, n = tok & 1023;
  float v[12], s = 0.f, s2 = 0.f;
#pragma unroll
  for (int i = 0; i < 12; ++i) {
    const int c = i * 32 + lane;
    const float t = x[((size_t)b * DIM + c) * NTOK + n];
    v[i] = t; s += t; s2 += t * t;
  }
#pragma unroll
  for (int off = 16; off; off >>= 1) {
    s  += __shfl_xor(s,  off, 32);
    s2 += __shfl_xor(s2, off, 32);
  }
  const float mu  = s * (1.f / DIM);
  const float var = s2 * (1.f / DIM) - mu * mu;
  const float r   = rsqrtf(var + LN_EPS);
  const size_t row = (size_t)tok * DIM;
#pragma unroll
  for (int i = 0; i < 12; ++i) {
    const int c = i * 32 + lane;
    xf[row + c] = v[i];
    xn[row + c] = (__bf16)((v[i] - mu) * r * g[c] + be[c]);
  }
}

// LN2 on the row-major residual x1.
__global__ void __launch_bounds__(256) ln2_k(const float* __restrict__ x1,
                                             const float* __restrict__ g,
                                             const float* __restrict__ be,
                                             __bf16* __restrict__ xn2) {
  const int tok  = (int)blockIdx.x * 8 + ((int)threadIdx.x >> 5);
  const int lane = threadIdx.x & 31;
  const size_t row = (size_t)tok * DIM;
  float v[12], s = 0.f, s2 = 0.f;
#pragma unroll
  for (int i = 0; i < 12; ++i) {
    const int c = i * 32 + lane;
    const float t = x1[row + c];
    v[i] = t; s += t; s2 += t * t;
  }
#pragma unroll
  for (int off = 16; off; off >>= 1) {
    s  += __shfl_xor(s,  off, 32);
    s2 += __shfl_xor(s2, off, 32);
  }
  const float mu  = s * (1.f / DIM);
  const float var = s2 * (1.f / DIM) - mu * mu;
  const float r   = rsqrtf(var + LN_EPS);
#pragma unroll
  for (int i = 0; i < 12; ++i) {
    const int c = i * 32 + lane;
    xn2[row + c] = (__bf16)((v[i] - mu) * r * g[c] + be[c]);
  }
}

// ---------------------------------------------------------------------------
// Generic GEMM: C[M,N] = epi(A[M,K] * W[N,K]^T). One wave -> 16x64 tile.
// All 4 waves of a block share n0, so the W k-slice (64x32 bf16 = 4KB) is
// staged once per block into LDS via async-to-LDS DMA.
// ---------------------------------------------------------------------------
struct QkvEpi {
  const float* bias; __bf16 *q, *k, *v;
  __device__ void operator()(int m, int n, float val) const {
    val += bias[n];
    const int part = n / DIM;
    const int oo   = n - part * DIM;
    const int h = oo / HEAD_DIM, d = oo - h * HEAD_DIM;
    const int b = m >> 10, seq = m & 1023;
    const size_t bh = (size_t)(b * HEADS + h);
    if (part == 0)      q[(bh * NTOK + seq) * HPAD + d] = (__bf16)(val * SCALE_F);
    else if (part == 1) k[(bh * NTOK + seq) * HPAD + d] = (__bf16)val;
    else                v[(bh * HPAD + d) * NTOK + seq] = (__bf16)val;  // V^T
  }
};

struct ProjEpi {
  const float* bias; const float* shortcut; float* x1;
  __device__ void operator()(int m, int n, float val) const {
    const size_t i = (size_t)m * DIM + n;
    x1[i] = val + bias[n] + shortcut[i];
  }
};

struct Fc1Epi {
  const float* bias; __bf16* h;
  __device__ void operator()(int m, int n, float val) const {
    const float xx = val + bias[n];
    const float ge = 0.5f * xx * (1.0f + erff(xx * 0.70710678118654752f));
    h[(size_t)m * HIDDEN + n] = (__bf16)ge;
  }
};

struct Fc2Epi {
  const float* bias; const float* x1; float* out;
  __device__ void operator()(int m, int n, float val) const {
    const float y = val + bias[n] + x1[(size_t)m * DIM + n];
    const int b = m >> 10, seq = m & 1023;
    out[((size_t)b * DIM + n) * NTOK + seq] = y;  // -> [B, C, H, W]
  }
};

template <typename Epi>
__global__ void __launch_bounds__(128) gemm16x64_k(const __bf16* __restrict__ A,
                                                   const __bf16* __restrict__ W,
                                                   int M, int N, int K, Epi epi) {
  __shared__ __bf16 wtile[64 * 32];            // 4KB W k-slice, block-shared
  const int tid  = (int)threadIdx.x;
  const int wv   = (int)blockIdx.x * 4 + (tid >> 5);
  const int mtiles = M >> 4;
  const int m0 = (wv % mtiles) << 4;
  const int n0 = (wv / mtiles) << 6;           // identical across the block
  const int lane = tid & 31;
  const int half = lane >> 4, ln = lane & 15;
  (void)N;

  v8f acc[4];
#pragma unroll
  for (int t = 0; t < 4; ++t) acc[t] = v8f_zero();

  for (int k = 0; k < K; k += 32) {
    // --- async-stage W[n0..n0+63][k..k+31] into LDS (256 x 16B chunks) ---
    for (int c = tid; c < 256; c += 128) {
      const int row  = c >> 2;                 // 0..63
      const int col8 = (c & 3) << 3;           // 0,8,16,24 (elements)
      const unsigned lds = (unsigned)(uintptr_t)(&wtile[row * 32 + col8]);
      const unsigned long long ga =
          (unsigned long long)(uintptr_t)(W + (size_t)(n0 + row) * K + k + col8);
      asm volatile("global_load_async_to_lds_b128 %0, %1, off"
                   :: "v"(lds), "v"(ga) : "memory");
    }
    asm volatile("s_wait_asynccnt 0" ::: "memory");
    __syncthreads();

    v16bf a = load_frag(A, K, m0, k);
#pragma unroll
    for (int t = 0; t < 4; ++t) {
      v16bf b;
      uint4* bu = reinterpret_cast<uint4*>(&b);
      const __bf16* rp = &wtile[(t * 16 + ln) * 32 + (half << 3)];
      bu[0] = *reinterpret_cast<const uint4*>(rp);
      bu[1] = *reinterpret_cast<const uint4*>(rp + 16);
      acc[t] = wmma_bf16(a, b, acc[t]);
    }
    __syncthreads();                           // before next overwrite
  }

#pragma unroll
  for (int t = 0; t < 4; ++t)
#pragma unroll
    for (int j = 0; j < 8; ++j)
      epi(m0 + j + 8 * half, n0 + t * 16 + ln, acc[t][j]);
}

// ---------------------------------------------------------------------------
// Transposed flash attention. One wave owns 16 query rows (on the lane axis).
//   S^T = K_tile x Q^T   (Q loaded once as loop-invariant B fragments)
//   softmax stats are per-lane scalars (1 shfl_xor(16) each for max & sum)
//   P^T fragment == the per-lane p values -> packed in registers, no LDS
//   O^T = V^T x P^T, accumulators rescale by a per-lane scalar alpha
// ---------------------------------------------------------------------------
__global__ void __launch_bounds__(256) flash_attn_k(const __bf16* __restrict__ qs,
                                                    const __bf16* __restrict__ ks,
                                                    const __bf16* __restrict__ vs,
                                                    __bf16* __restrict__ outp) {
  const int wv   = (int)blockIdx.x * 8 + ((int)threadIdx.x >> 5);
  const int lane = threadIdx.x & 31;
  const int half = lane >> 4, ln = lane & 15;
  const int bh = wv >> 6;
  const int m0 = (wv & 63) << 4;
  const __bf16* qp = qs + (size_t)bh * (NTOK * HPAD);
  const __bf16* kp = ks + (size_t)bh * (NTOK * HPAD);
  const __bf16* vp = vs + (size_t)bh * (NTOK * HPAD);

  // Q as B fragments (lane = query column), loop-invariant.
  const v16bf qb0 = load_frag(qp, HPAD, m0, 0);
  const v16bf qb1 = load_frag(qp, HPAD, m0, 32);

  float rmax = -1e30f, rsum = 0.f;
  v8f o[3];
#pragma unroll
  for (int t = 0; t < 3; ++t) o[t] = v8f_zero();

  for (int nb = 0; nb < NTOK; nb += 32) {
    // S^T tiles: rows = keys (VGPR axis), cols = queries (lane axis)
    v8f st0 = v8f_zero(), st1 = v8f_zero();
    st0 = wmma_bf16(load_frag(kp, HPAD, nb,      0),  qb0, st0);
    st0 = wmma_bf16(load_frag(kp, HPAD, nb,      32), qb1, st0);
    st1 = wmma_bf16(load_frag(kp, HPAD, nb + 16, 0),  qb0, st1);
    st1 = wmma_bf16(load_frag(kp, HPAD, nb + 16, 32), qb1, st1);

    // per-query (per-lane) running max over the 32 keys of this block
    float tm = fmaxf(st0[0], st1[0]);
#pragma unroll
    for (int j = 1; j < 8; ++j) tm = fmaxf(tm, fmaxf(st0[j], st1[j]));
    tm = fmaxf(tm, __shfl_xor(tm, 16, 32));
    const float nm = fmaxf(rmax, tm);
    const float al = __expf(rmax - nm);
    rmax = nm;

    float p0[8], p1[8], ts = 0.f;
#pragma unroll
    for (int j = 0; j < 8; ++j) {
      p0[j] = __expf(st0[j] - nm);
      p1[j] = __expf(st1[j] - nm);
      ts += p0[j] + p1[j];
    }
    ts += __shfl_xor(ts, 16, 32);
    rsum = rsum * al + ts;

    // P^T B-fragment straight from registers:
    // lane holds query=ln; its keys are exactly K chunks {8h..8h+7, 16+8h..}
    v16bf pb;
#pragma unroll
    for (int j = 0; j < 8; ++j) {
      pb[j]     = (__bf16)p0[j];
      pb[j + 8] = (__bf16)p1[j];
    }

#pragma unroll
    for (int t = 0; t < 3; ++t) {
      o[t] = o[t] * al;                                  // per-lane rescale
      o[t] = wmma_bf16(load_frag(vp, NTOK, t * 16, nb), pb, o[t]);
    }
  }

  // O^T tile: row = dim (VGPR axis), col = query (lane axis). Dims j+8h are
  // contiguous -> 16B packed stores.
  const float inv = 1.f / rsum;
  const int b = bh >> 3, h = bh & 7;
  const int seq = m0 + ln;
  __bf16* obase = outp + (size_t)(b * NTOK + seq) * DIM + h * HEAD_DIM;
#pragma unroll
  for (int t = 0; t < 3; ++t) {
    union { __bf16 e[8]; uint4 u; } pk;
#pragma unroll
    for (int j = 0; j < 8; ++j) pk.e[j] = (__bf16)(o[t][j] * inv);
    *reinterpret_cast<uint4*>(obase + t * 16 + (half << 3)) = pk.u;
  }
}

// ---------------------------------------------------------------------------
// Host-side orchestration
// ---------------------------------------------------------------------------
extern "C" void kernel_launch(void* const* d_in, const int* in_sizes, int n_in,
                              void* d_out, int out_size, void* d_ws, size_t ws_size,
                              hipStream_t stream) {
  (void)in_sizes; (void)n_in; (void)out_size; (void)ws_size;
  const float* x      = (const float*)d_in[0];
  const float* w_qkv  = (const float*)d_in[1];
  const float* b_qkv  = (const float*)d_in[2];
  const float* w_proj = (const float*)d_in[3];
  const float* b_proj = (const float*)d_in[4];
  const float* g1     = (const float*)d_in[5];
  const float* be1    = (const float*)d_in[6];
  const float* g2     = (const float*)d_in[7];
  const float* be2    = (const float*)d_in[8];
  const float* w_fc1  = (const float*)d_in[9];
  const float* b_fc1  = (const float*)d_in[10];
  const float* w_fc2  = (const float*)d_in[11];
  const float* b_fc2  = (const float*)d_in[12];
  float* out = (float*)d_out;

  char* p = (char*)d_ws;
  auto take = [&](size_t bytes) -> char* {
    char* r = p;
    p += (bytes + 255) & ~(size_t)255;
    return r;
  };
  float*  xf     = (float*) take((size_t)MTOT * DIM * 4);
  float*  x1     = (float*) take((size_t)MTOT * DIM * 4);
  __bf16* xn     = (__bf16*)take((size_t)MTOT * DIM * 2);
  __bf16* xn2    = (__bf16*)take((size_t)MTOT * DIM * 2);
  __bf16* attn_o = (__bf16*)take((size_t)MTOT * DIM * 2);
  __bf16* hs     = (__bf16*)take((size_t)MTOT * HIDDEN * 2);
  const size_t stg = (size_t)BATCH * HEADS * NTOK * HPAD;   // 4M elems
  __bf16* qs     = (__bf16*)take(stg * 2);                  // contiguous q/k/v
  __bf16* kk     = (__bf16*)take(stg * 2);
  __bf16* vs     = (__bf16*)take(stg * 2);
  __bf16* wqkvb  = (__bf16*)take((size_t)3 * DIM * DIM * 2);
  __bf16* wprojb = (__bf16*)take((size_t)DIM * DIM * 2);
  __bf16* wfc1b  = (__bf16*)take((size_t)HIDDEN * DIM * 2);
  __bf16* wfc2b  = (__bf16*)take((size_t)DIM * HIDDEN * 2);

  // Zero q/k/v staging (dim padding 48..63 must be 0 for the WMMA K-loop).
  {
    const size_t n16 = 3 * stg * 2 / 16;    // 1,572,864 uint4
    zero16_k<<<(unsigned)(n16 / 256), 256, 0, stream>>>((uint4*)qs);
  }

  // Weights -> bf16
  auto cvt = [&](const float* src, __bf16* dst, int n) {
    cvt_bf16_k<<<(n + 255) / 256, 256, 0, stream>>>(src, dst, n);
  };
  cvt(w_qkv,  wqkvb,  3 * DIM * DIM);
  cvt(w_proj, wprojb, DIM * DIM);
  cvt(w_fc1,  wfc1b,  HIDDEN * DIM);
  cvt(w_fc2,  wfc2b,  DIM * HIDDEN);

  // 1) transpose + LN1
  prep_ln1_k<<<MTOT / 8, 256, 0, stream>>>(x, g1, be1, xf, xn);

  // 2) QKV projection -> per-head staging (q pre-scaled, V transposed)
  {
    const int waves = (MTOT / 16) * ((3 * DIM) / 64);       // 9216
    gemm16x64_k<<<waves / 4, 128, 0, stream>>>(xn, wqkvb, MTOT, 3 * DIM, DIM,
                                               QkvEpi{b_qkv, qs, kk, vs});
  }

  // 3) flash attention -> attn_o [token, C]
  flash_attn_k<<<(BATCH * HEADS * 64) / 8, 256, 0, stream>>>(qs, kk, vs, attn_o);

  // 4) output projection + residual -> x1 (fp32)
  {
    const int waves = (MTOT / 16) * (DIM / 64);             // 3072
    gemm16x64_k<<<waves / 4, 128, 0, stream>>>(attn_o, wprojb, MTOT, DIM, DIM,
                                               ProjEpi{b_proj, xf, x1});
  }

  // 5) LN2
  ln2_k<<<MTOT / 8, 256, 0, stream>>>(x1, g2, be2, xn2);

  // 6) FC1 + exact GELU
  {
    const int waves = (MTOT / 16) * (HIDDEN / 64);          // 12288
    gemm16x64_k<<<waves / 4, 128, 0, stream>>>(xn2, wfc1b, MTOT, HIDDEN, DIM,
                                               Fc1Epi{b_fc1, hs});
  }

  // 7) FC2 + residual + [token,C] -> [B,C,H,W] store
  {
    const int waves = (MTOT / 16) * (DIM / 64);             // 3072
    gemm16x64_k<<<waves / 4, 128, 0, stream>>>(hs, wfc2b, MTOT, DIM, HIDDEN,
                                               Fc2Epi{b_fc2, x1, out});
  }
}
